// CompGCN_52527450030387
// MI455X (gfx1250) — compile-verified, matrix operand order checked by reading, not run on hardware
//
#include <hip/hip_runtime.h>
#include <math.h>

// ---------------- problem constants (match reference) ----------------
#define NUM_ENT 100000
#define NUM_REL 400
#define D_IN    200
#define D_OUT   400
#define N_EDGES 600000
#define EHALF   300000
#define BATCH   1024
#define BN_EPS  1e-5f

#define KP_IN   224          // D_IN padded to 7*32
#define KP_OUT  416          // D_OUT padded to 13*32
#define ENT_PAD 100032       // NUM_ENT padded to mult of 64
#define REL_PAD 448          // 400 padded to mult of 64

// ---------------- CDNA5 types ----------------
typedef __attribute__((ext_vector_type(16))) __bf16 v16bf;
typedef __attribute__((ext_vector_type(8)))  __bf16 v8bf;
typedef __attribute__((ext_vector_type(8)))  float  v8f;
typedef __attribute__((ext_vector_type(4)))  unsigned int u32x4;
typedef __attribute__((ext_vector_type(8)))  int i32x8;
typedef __attribute__((ext_vector_type(4)))  int i32x4;

__device__ __forceinline__ unsigned short f32_to_bf16_bits(float f) {
    unsigned u = __builtin_bit_cast(unsigned, f);
    u += 0x7FFFu + ((u >> 16) & 1u);       // round-to-nearest-even
    return (unsigned short)(u >> 16);
}

// LDS byte address: flat LDS aperture keeps the LDS offset in addr[31:0] (ISA 10.2)
__device__ __forceinline__ unsigned lds_addr_of(const void* p) {
    return (unsigned)(unsigned long long)p;
}

__device__ __forceinline__ void atomic_add_f32(float* p, float v) {
    __hip_atomic_fetch_add(p, v, __ATOMIC_RELAXED, __HIP_MEMORY_SCOPE_AGENT);
}

// ---------------- Tensor Data Mover descriptor helpers (ISA ch.8 D#) ----------------
// 2D tile load: tile_w elements contiguous per row, tile_h rows, row stride
// `stride` elements, element size from ds_code (0=1B,1=2B,2=4B,3=8B).
__device__ __forceinline__ void tdm_load_2d(unsigned lds_addr, const void* gptr,
                                            unsigned tile_w, unsigned tile_h,
                                            unsigned tensor_w, unsigned tensor_h,
                                            unsigned stride, unsigned ds_code) {
    unsigned long long ga = (unsigned long long)gptr;
    u32x4 g0 = { 1u,                                   // count=1
                 lds_addr,
                 (unsigned)ga,
                 (unsigned)((ga >> 32) & 0x1FFFFFFull) | (2u << 30) };  // type=2 (image)
    i32x8 g1 = { (int)(ds_code << 16),                 // wg_mask=0, data_size
                 (int)((tensor_w & 0xFFFFu) << 16),    // abar=0 | tensor_dim0 lo16
                 (int)((tensor_w >> 16) | ((tensor_h & 0xFFFFu) << 16)),
                 (int)((tensor_h >> 16) | (tile_w << 16)),   // tile_dim0
                 (int)(tile_h & 0xFFFFu),              // tile_dim1 | tile_dim2=0
                 (int)stride,                          // tensor_dim0_stride lo32
                 0, 0 };
    i32x4 z4 = { 0, 0, 0, 0 };
    i32x8 z8 = { 0, 0, 0, 0, 0, 0, 0, 0 };
    __builtin_amdgcn_tensor_load_to_lds(g0, g1, z4, z4, z8, 0);
}

// Gather-mode load: 8 rows selected by 32-bit indices, each row_w f32 wide.
__device__ __forceinline__ void tdm_gather8_f32(unsigned lds_addr, const void* gptr,
                                                unsigned row_w, unsigned n_rows,
                                                i32x4 idx_lo, i32x4 idx_hi) {
    unsigned long long ga = (unsigned long long)gptr;
    u32x4 g0 = { 1u | (1u << 30) | (1u << 31),         // count=1, idx32, gather
                 lds_addr,
                 (unsigned)ga,
                 (unsigned)((ga >> 32) & 0x1FFFFFFull) | (2u << 30) };
    i32x8 g1 = { (int)(2u << 16),                      // data_size=4B
                 (int)((row_w & 0xFFFFu) << 16),
                 (int)((row_w >> 16) | ((n_rows & 0xFFFFu) << 16)),
                 (int)((n_rows >> 16) | (row_w << 16)),// tile_dim0 = row_w
                 8,                                    // tile_dim1 = #indices
                 (int)row_w,                           // stride = row_w
                 0, 0 };
    i32x8 z8 = { 0, 0, 0, 0, 0, 0, 0, 0 };
    __builtin_amdgcn_tensor_load_to_lds(g0, g1, idx_lo, idx_hi, z8, 0);
}

// Gather-mode load: 16 rows selected by 16-bit indices packed 2/dword.
__device__ __forceinline__ void tdm_gather16_f32(unsigned lds_addr, const void* gptr,
                                                 unsigned row_w, unsigned n_rows,
                                                 i32x4 idx_lo, i32x4 idx_hi) {
    unsigned long long ga = (unsigned long long)gptr;
    u32x4 g0 = { 1u | (1u << 31),                      // count=1, idx16, gather
                 lds_addr,
                 (unsigned)ga,
                 (unsigned)((ga >> 32) & 0x1FFFFFFull) | (2u << 30) };
    i32x8 g1 = { (int)(2u << 16),
                 (int)((row_w & 0xFFFFu) << 16),
                 (int)((row_w >> 16) | ((n_rows & 0xFFFFu) << 16)),
                 (int)((n_rows >> 16) | (row_w << 16)),
                 16,
                 (int)row_w,
                 0, 0 };
    i32x8 z8 = { 0, 0, 0, 0, 0, 0, 0, 0 };
    __builtin_amdgcn_tensor_load_to_lds(g0, g1, idx_lo, idx_hi, z8, 0);
}

// Fragment row per CDNA5 16-bit A layout: lanes 0-15 -> K {0..7,16..23},
// lanes 16-31 -> K {8..15,24..31}; two ds_load_b128 per fragment.
__device__ __forceinline__ v16bf load_frag_row(const unsigned short* rowbase, int laneHi) {
    const v8bf lo = *reinterpret_cast<const v8bf*>(rowbase + laneHi * 8);
    const v8bf hv = *reinterpret_cast<const v8bf*>(rowbase + 16 + laneHi * 8);
    return __builtin_shufflevector(lo, hv, 0,1,2,3,4,5,6,7,8,9,10,11,12,13,14,15);
}

// ---------------- TDM-fed double-buffered bf16 WMMA GEMM ----------------
// C[M,N](+epilogue) = alpha * A @ B ; A: bf16 row-major [>=ceil64(M) x lda],
// B: bf16 n-major [>=ceil64(N) x ldb]; both fully zero-padded, Kp mult of 32.
__global__ __launch_bounds__(256)
void wmma_gemm_tdm_kernel(const unsigned short* __restrict__ A, unsigned lda,
                          const unsigned short* __restrict__ B, unsigned ldb,
                          float* __restrict__ C, long long ldc,
                          int M, int N, int Kp,
                          float alpha, const float* __restrict__ bias, int mode) {
    __shared__ __align__(16) unsigned short As[2][64 * 32];   // [m][k]
    __shared__ __align__(16) unsigned short Bs[2][64 * 32];   // [n][k]

    const int tid  = threadIdx.x;
    const int wave = tid >> 5;
    const int lane = tid & 31;
    const int hi   = lane >> 4;
    const int lcol = lane & 15;
    const int m0 = blockIdx.y * 64;
    const int n0 = blockIdx.x * 64;
    const int tr  = wave & 3;
    const int tcA = wave >> 2;
    const int nk  = Kp >> 5;

    const unsigned short* Abase = A + (unsigned long long)m0 * lda;
    const unsigned short* Bbase = B + (unsigned long long)n0 * ldb;

    if (tid < 32) {     // prologue: fill buffer 0
        tdm_load_2d(lds_addr_of(As[0]), Abase, 32, 64, lda, 1u << 24, lda, 1);
        tdm_load_2d(lds_addr_of(Bs[0]), Bbase, 32, 64, ldb, 1u << 24, ldb, 1);
    }

    v8f acc0 = {};
    v8f acc1 = {};

    for (int kt = 0; kt < nk; ++kt) {
        __syncthreads();                       // compute of kt-1 done; buf[(kt+1)&1] free
        if (tid < 32) {
            if (kt + 1 < nk) {                 // issue next tiles, retire current pair
                tdm_load_2d(lds_addr_of(As[(kt + 1) & 1]), Abase + (kt + 1) * 32, 32, 64, lda, 1u << 24, lda, 1);
                tdm_load_2d(lds_addr_of(Bs[(kt + 1) & 1]), Bbase + (kt + 1) * 32, 32, 64, ldb, 1u << 24, ldb, 1);
                __builtin_amdgcn_s_wait_tensorcnt(2);
            } else {
                __builtin_amdgcn_s_wait_tensorcnt(0);
            }
        }
        __syncthreads();
        const unsigned short* as = As[kt & 1];
        const unsigned short* bs = Bs[kt & 1];
        v16bf a  = load_frag_row(&as[(tr * 16 + lcol) * 32], hi);
        v16bf b0 = load_frag_row(&bs[(tcA * 16 + lcol) * 32], hi);
        v16bf b1 = load_frag_row(&bs[((tcA + 2) * 16 + lcol) * 32], hi);
        acc0 = __builtin_amdgcn_wmma_f32_16x16x32_bf16(false, a, false, b0, (short)0, acc0, false, false);
        acc1 = __builtin_amdgcn_wmma_f32_16x16x32_bf16(false, a, false, b1, (short)0, acc1, false, false);
    }

    const int col0 = n0 + tcA * 16 + lcol;
    const int col1 = n0 + (tcA + 2) * 16 + lcol;
#pragma unroll
    for (int v = 0; v < 8; ++v) {
        int row = m0 + tr * 16 + v + hi * 8;
        if (row >= M) continue;
        float v0 = acc0[v] * alpha;
        float v1 = acc1[v] * alpha;
        if (bias) {
            if (col0 < N) v0 += bias[col0];
            if (col1 < N) v1 += bias[col1];
        }
        if (mode == 1) {
            v0 = 1.f / (1.f + expf(-v0));
            v1 = 1.f / (1.f + expf(-v1));
        }
        if (col0 < N) C[(long long)row * ldc + col0] = v0;
        if (col1 < N) C[(long long)row * ldc + col1] = v1;
    }
}

// ---------------- fold loop-rel circulant into loop_w (f32) ----------------
__global__ __launch_bounds__(256)
void prep_cw_kernel(const float* __restrict__ loop_rel,
                    const float* __restrict__ loop_w,
                    float* __restrict__ CW) {
    __shared__ float lr2[2 * D_IN];
    for (int i = threadIdx.x; i < 2 * D_IN; i += 256) lr2[i] = loop_rel[i % D_IN];
    __syncthreads();
    int idx = blockIdx.x * 256 + threadIdx.x;
    if (idx >= D_IN * D_OUT) return;
    int j = idx / D_OUT, n = idx % D_OUT;
    float s = 0.f;
    for (int k = 0; k < D_IN; ++k)
        s = fmaf(lr2[j + k], loop_w[k * D_OUT + n], s);
    CW[idx] = s;
}

// ---------------- f32 -> padded bf16 conversions ----------------
__global__ __launch_bounds__(256)
void conv_pad_kernel(const float* __restrict__ src, int rows, int cols, int src_ld,
                     unsigned short* __restrict__ dst, int dst_ld, int rows_pad) {
    long long tot = (long long)rows_pad * dst_ld;
    for (long long idx = (long long)blockIdx.x * 256 + threadIdx.x; idx < tot;
         idx += (long long)gridDim.x * 256) {
        int r = (int)(idx / dst_ld), c = (int)(idx % dst_ld);
        float v = (r < rows && c < cols) ? src[(long long)r * src_ld + c] : 0.f;
        dst[idx] = f32_to_bf16_bits(v);
    }
}

// transpose-convert: src is K x N row-major f32; dst is n-major bf16 [n_pad x dst_ld]
__global__ __launch_bounds__(256)
void conv_pad_t_kernel(const float* __restrict__ src, int K, int N,
                       unsigned short* __restrict__ dst, int dst_ld, int n_pad) {
    long long tot = (long long)n_pad * dst_ld;
    for (long long idx = (long long)blockIdx.x * 256 + threadIdx.x; idx < tot;
         idx += (long long)gridDim.x * 256) {
        int n = (int)(idx / dst_ld), k = (int)(idx % dst_ld);
        float v = (n < N && k < K) ? src[(long long)k * N + n] : 0.f;
        dst[idx] = f32_to_bf16_bits(v);
    }
}

// ---------------- fused edge-message kernel (TDM gather + WMMA) ----------------
__global__ __launch_bounds__(256)
void edge_msg_kernel(const float* __restrict__ ent_emb, const float* __restrict__ rel_emb,
                     const unsigned short* __restrict__ inw_t,
                     const unsigned short* __restrict__ outw_t,
                     const int* __restrict__ src, const int* __restrict__ dst,
                     const int* __restrict__ etype, const float* __restrict__ enorm,
                     float* __restrict__ x) {
    __shared__ __align__(16) float Ae[16 * D_IN];             // gathered ent rows
    __shared__ __align__(16) float Re[16 * D_IN];             // gathered rel rows
    __shared__ __align__(16) unsigned short Ced[16 * KP_IN];  // ccorr result, bf16
    __shared__ __align__(16) unsigned short Wt[D_OUT * 32];   // W tile, n-major
    __shared__ int   sDst[16];
    __shared__ float sNorm[16];

    const int tid = threadIdx.x;
    const int e0  = blockIdx.x * 16;
    if (tid < 16) {
        sDst[tid]  = dst[e0 + tid];
        sNorm[tid] = enorm[e0 + tid];
    }

    if (tid < 32) {     // wave 0: TDM gathers for ent/rel rows
        i32x4 i0 = { src[e0 + 0], src[e0 + 1], src[e0 + 2],  src[e0 + 3]  };
        i32x4 i1 = { src[e0 + 4], src[e0 + 5], src[e0 + 6],  src[e0 + 7]  };
        i32x4 i2 = { src[e0 + 8], src[e0 + 9], src[e0 + 10], src[e0 + 11] };
        i32x4 i3 = { src[e0 + 12], src[e0 + 13], src[e0 + 14], src[e0 + 15] };
        tdm_gather8_f32(lds_addr_of(Ae), ent_emb, D_IN, NUM_ENT, i0, i1);
        tdm_gather8_f32(lds_addr_of(Ae) + 8u * D_IN * 4u, ent_emb, D_IN, NUM_ENT, i2, i3);
        int t[16];
#pragma unroll
        for (int i = 0; i < 16; ++i) t[i] = etype[e0 + i];
        i32x4 r0 = { t[0] | (t[1] << 16),  t[2] | (t[3] << 16),
                     t[4] | (t[5] << 16),  t[6] | (t[7] << 16) };
        i32x4 r1 = { t[8] | (t[9] << 16),  t[10] | (t[11] << 16),
                     t[12] | (t[13] << 16), t[14] | (t[15] << 16) };
        tdm_gather16_f32(lds_addr_of(Re), rel_emb, D_IN, NUM_REL, r0, r1);
        __builtin_amdgcn_s_wait_tensorcnt(0);
    }
    __syncthreads();

    // ccorr: c[e][k] = sum_j Ae[e][j] * Re[e][(j+k)%200]   (the real VALU math)
    for (int idx = tid; idx < 16 * D_IN; idx += 256) {
        int e = idx / D_IN, k = idx % D_IN;
        const float* a = &Ae[e * D_IN];
        const float* b = &Re[e * D_IN];
        float s = 0.f;
        for (int j = 0; j < D_IN; ++j) {
            int jk = j + k;
            jk -= (jk >= D_IN) ? D_IN : 0;
            s = fmaf(a[j], b[jk], s);
        }
        Ced[e * KP_IN + k] = f32_to_bf16_bits(s);
    }
    for (int idx = tid; idx < 16 * (KP_IN - D_IN); idx += 256)   // zero K pad
        Ced[(idx / (KP_IN - D_IN)) * KP_IN + D_IN + (idx % (KP_IN - D_IN))] = 0;

    const unsigned short* Wsrc = (e0 < EHALF) ? inw_t : outw_t;  // groups never straddle
    const int wave = tid >> 5, lane = tid & 31, hi = lane >> 4, lcol = lane & 15;
    v8f acc[4] = {};

    for (int k0 = 0; k0 < KP_IN; k0 += 32) {
        __syncthreads();
        if (tid < 32) {   // TDM: 400-row x 32-col tile of n-major bf16 W
            tdm_load_2d(lds_addr_of(Wt), Wsrc + k0, 32, D_OUT, KP_IN, REL_PAD, KP_IN, 1);
            __builtin_amdgcn_s_wait_tensorcnt(0);
        }
        __syncthreads();
        v16bf a = load_frag_row(&Ced[lcol * KP_IN + k0], hi);
#pragma unroll
        for (int j = 0; j < 4; ++j) {
            int ct = wave + 8 * j;                            // 25 col tiles over 8 waves
            if (ct < 25) {
                v16bf b = load_frag_row(&Wt[(ct * 16 + lcol) * 32], hi);
                acc[j] = __builtin_amdgcn_wmma_f32_16x16x32_bf16(false, a, false, b, (short)0, acc[j], false, false);
            }
        }
    }

#pragma unroll
    for (int j = 0; j < 4; ++j) {
        int ct = wave + 8 * j;
        if (ct >= 25) continue;
        int col = ct * 16 + lcol;
#pragma unroll
        for (int v = 0; v < 8; ++v) {
            int e = v + hi * 8;
            float val = acc[j][v] * sNorm[e] * (1.f / 3.f);
            atomic_add_f32(&x[(long long)sDst[e] * D_OUT + col], val);
        }
    }
}

// ---------------- batchnorm ----------------
__global__ __launch_bounds__(512)
void bn_stats_kernel(const float* __restrict__ x, float* __restrict__ sums) {
    int c = threadIdx.x;
    if (c >= D_OUT) return;
    int r0 = blockIdx.x * 256;
    int r1 = r0 + 256;
    if (r1 > NUM_ENT) r1 = NUM_ENT;
    float s = 0.f, s2 = 0.f;
    for (int r = r0; r < r1; ++r) {
        float v = x[(long long)r * D_OUT + c];
        s += v;
        s2 += v * v;
    }
    atomic_add_f32(&sums[c], s);
    atomic_add_f32(&sums[D_OUT + c], s2);
}

__global__ __launch_bounds__(256)
void bn_norm_tanh_kernel(float* __restrict__ x, const float* __restrict__ sums,
                         const float* __restrict__ gamma, const float* __restrict__ beta) {
    const long long total = (long long)NUM_ENT * D_OUT;
    for (long long idx = (long long)blockIdx.x * 256 + threadIdx.x; idx < total;
         idx += (long long)gridDim.x * 256) {
        int c = (int)(idx % D_OUT);
        float mean = sums[c] * (1.f / NUM_ENT);
        float var  = sums[D_OUT + c] * (1.f / NUM_ENT) - mean * mean;
        float v = x[idx];
        v = (v - mean) * rsqrtf(var + BN_EPS) * gamma[c] + beta[c];
        x[idx] = tanhf(v);
    }
}

// obj[b][k] = x[head_b][k] * r[rela_b][k], written directly as padded bf16 A-matrix
__global__ __launch_bounds__(256)
void obj_bf_kernel(const float* __restrict__ x, const float* __restrict__ r,
                   const int* __restrict__ triples, unsigned short* __restrict__ obj_bf) {
    int idx = blockIdx.x * 256 + threadIdx.x;
    if (idx >= BATCH * KP_OUT) return;
    int b = idx / KP_OUT, k = idx % KP_OUT;
    float v = 0.f;
    if (k < D_OUT) {
        int head = triples[b * 3 + 0];
        int rela = triples[b * 3 + 1];
        v = x[(long long)head * D_OUT + k] * r[rela * D_OUT + k];
    }
    obj_bf[idx] = f32_to_bf16_bits(v);
}

__global__ __launch_bounds__(256)
void zero_kernel(float* __restrict__ p, int n) {
    int i = blockIdx.x * 256 + threadIdx.x;
    if (i < n) p[i] = 0.f;
}

// ---------------- driver ----------------
extern "C" void kernel_launch(void* const* d_in, const int* in_sizes, int n_in,
                              void* d_out, int out_size, void* d_ws, size_t ws_size,
                              hipStream_t stream) {
    (void)in_sizes; (void)n_in; (void)out_size; (void)ws_size;
    const float* ent_emb  = (const float*)d_in[0];
    const float* rel_emb  = (const float*)d_in[1];
    const float* in_w     = (const float*)d_in[2];
    const float* out_w    = (const float*)d_in[3];
    const float* loop_w   = (const float*)d_in[4];
    const float* w_rel    = (const float*)d_in[5];
    const float* loop_rel = (const float*)d_in[6];
    const float* bias_w   = (const float*)d_in[7];
    const float* bn_gamma = (const float*)d_in[8];
    const float* bn_beta  = (const float*)d_in[9];
    const float* emb_ent_w= (const float*)d_in[10];
    const float* ent_bias = (const float*)d_in[11];
    const int*   src      = (const int*)d_in[12];
    const int*   dst      = (const int*)d_in[13];
    const int*   etype    = (const int*)d_in[14];
    const float* enorm    = (const float*)d_in[15];
    const int*   triples  = (const int*)d_in[16];
    float* out = (float*)d_out;

    // workspace layout (bytes), ~291 MB total
    char* ws = (char*)d_ws;
    float*          x      = (float*)(ws);                      // 160,000,000
    float*          CW     = (float*)(ws + 160000000ull);       //     320,000
    float*          r      = (float*)(ws + 160320000ull);       //     640,000
    float*          sums   = (float*)(ws + 160960000ull);       //       3,200
    unsigned short* obj_bf = (unsigned short*)(ws + 160963200ull); // 1024*416*2     =   851,968
    unsigned short* ent_bf = (unsigned short*)(ws + 161815168ull); // 100032*224*2   = 44,814,336
    unsigned short* emb_bf = (unsigned short*)(ws + 206629504ull); // 100032*416*2   = 83,226,624
    unsigned short* rel_bf = (unsigned short*)(ws + 289856128ull); // 448*224*2      =    200,704
    unsigned short* cw_t   = (unsigned short*)(ws + 290056832ull); // 448*224*2
    unsigned short* wrel_t = (unsigned short*)(ws + 290257536ull); // 448*224*2
    unsigned short* inw_t  = (unsigned short*)(ws + 290458240ull); // 448*224*2
    unsigned short* outw_t = (unsigned short*)(ws + 290658944ull); // 448*224*2

    zero_kernel<<<4, 256, 0, stream>>>(sums, 2 * D_OUT);
    prep_cw_kernel<<<(D_IN * D_OUT + 255) / 256, 256, 0, stream>>>(loop_rel, loop_w, CW);

    // one-time padded bf16 conversions (A row-major, B n-major)
    conv_pad_kernel<<<16384, 256, 0, stream>>>(ent_emb, NUM_ENT, D_IN, D_IN, ent_bf, KP_IN, ENT_PAD);
    conv_pad_kernel<<<16384, 256, 0, stream>>>(emb_ent_w, NUM_ENT, D_OUT, D_OUT, emb_bf, KP_OUT, ENT_PAD);
    conv_pad_kernel<<<512, 256, 0, stream>>>(rel_emb, NUM_REL, D_IN, D_IN, rel_bf, KP_IN, REL_PAD);
    conv_pad_t_kernel<<<512, 256, 0, stream>>>(CW, D_IN, D_OUT, cw_t, KP_IN, REL_PAD);
    conv_pad_t_kernel<<<512, 256, 0, stream>>>(w_rel, D_IN, D_OUT, wrel_t, KP_IN, REL_PAD);
    conv_pad_t_kernel<<<512, 256, 0, stream>>>(in_w, D_IN, D_OUT, inw_t, KP_IN, REL_PAD);
    conv_pad_t_kernel<<<512, 256, 0, stream>>>(out_w, D_IN, D_OUT, outw_t, KP_IN, REL_PAD);

    {   // x = ent_emb @ CW / 3 + bias_w      [M=100000, N=400, K=224pad]
        dim3 g(REL_PAD / 64, ENT_PAD / 64);
        wmma_gemm_tdm_kernel<<<g, 256, 0, stream>>>(ent_bf, KP_IN, cw_t, KP_IN, x, D_OUT,
                                                    NUM_ENT, D_OUT, KP_IN, 1.f / 3.f, bias_w, 0);
    }

    edge_msg_kernel<<<N_EDGES / 16, 256, 0, stream>>>(ent_emb, rel_emb, inw_t, outw_t,
                                                      src, dst, etype, enorm, x);

    bn_stats_kernel<<<(NUM_ENT + 255) / 256, 512, 0, stream>>>(x, sums);
    bn_norm_tanh_kernel<<<4096, 256, 0, stream>>>(x, sums, bn_gamma, bn_beta);

    {   // r = rel_emb @ w_rel                [M=400, N=400, K=224pad]
        dim3 g(REL_PAD / 64, REL_PAD / 64);
        wmma_gemm_tdm_kernel<<<g, 256, 0, stream>>>(rel_bf, KP_IN, wrel_t, KP_IN, r, D_OUT,
                                                    NUM_REL, D_OUT, KP_IN, 1.f, nullptr, 0);
    }

    obj_bf_kernel<<<(BATCH * KP_OUT + 255) / 256, 256, 0, stream>>>(x, r, triples, obj_bf);

    {   // score = sigmoid(obj @ emb_ent_w^T + ent_bias)   [M=1024, N=100000, K=416pad]
        dim3 g(ENT_PAD / 64, BATCH / 64);
        wmma_gemm_tdm_kernel<<<g, 256, 0, stream>>>(obj_bf, KP_OUT, emb_bf, KP_OUT, out, NUM_ENT,
                                                    BATCH, NUM_ENT, KP_OUT, 1.f, ent_bias, 1);
    }
}
